// MultiheadAttention_26903675142513
// MI455X (gfx1250) — matrix operand order repaired and checked by reading, not compiled, not run on hardware
//
#include <hip/hip_runtime.h>
#include <hip/hip_bf16.h>

// MultiheadAttention for MI455X (gfx1250), bf16 WMMA path.
// Pipeline: f32->bf16 convert, W transpose, QKV GEMM (async-LDS staged B tile,
// scatter to Q/K row-major + V transposed), per-(b,h,qtile) flash attention,
// output GEMM (async-LDS staged B tile, fp32 out).
// GEMMs: 64x64 block tile, 4 waves share one B tile double-buffered in LDS via
// global_load_async_to_lds_b128 (ASYNCcnt pipeline). Working set ~120 MB is
// L2-resident (192 MB L2), so A-fragments stream straight from L2.

typedef __attribute__((ext_vector_type(16))) __bf16 v16bf;
typedef __attribute__((ext_vector_type(8)))  float  v8f;
typedef unsigned short u16;
typedef unsigned int   u32;
typedef unsigned long long u64;

#define BDIM 4
#define LDIM 2048
#define EDIM 1024
#define HDIM 16
#define ADIM 64
#define E3   3072
#define LDST 72          // LDS row stride in u16 (144B: 16B aligned, bank-spread)

union VB {
  v16bf v;
  uint4 q[2];
  u32   u[8];
};

__device__ __forceinline__ u16 f2bf(float f) {
  u32 u = __builtin_bit_cast(u32, f);
  u32 r = u + 0x7FFFu + ((u >> 16) & 1u);   // round-to-nearest-even
  return (u16)(r >> 16);
}
__device__ __forceinline__ u32 pack2(float a, float b) {
  return (u32)f2bf(a) | ((u32)f2bf(b) << 16);
}
__device__ __forceinline__ uint4 ld16(const u16* p) {
  return *reinterpret_cast<const uint4*>(p);
}
__device__ __forceinline__ v8f vzero() {
  v8f z;
#pragma unroll
  for (int i = 0; i < 8; ++i) z[i] = 0.f;
  return z;
}
__device__ __forceinline__ v8f wmma_bf16(const VB& a, const VB& b, v8f c) {
  return __builtin_amdgcn_wmma_f32_16x16x32_bf16(false, a.v, false, b.v,
                                                 (short)0, c, false, false);
}

// Async DMA: 16B from global (per-lane addr) into LDS (per-lane byte offset).
__device__ __forceinline__ void async_ld_b128(u32 lds_byte_off, const void* gptr) {
  asm volatile("global_load_async_to_lds_b128 %0, %1, off"
               :: "v"(lds_byte_off), "v"((u64)(size_t)gptr) : "memory");
}
__device__ __forceinline__ void wait_async0() {
  asm volatile("s_wait_asynccnt 0x0" ::: "memory");
}

// ---------------- elementwise f32 -> bf16 ----------------
__global__ __launch_bounds__(256) void k_f2bf(const float* __restrict__ in,
                                              u16* __restrict__ out, int n) {
  int i = blockIdx.x * 256 + threadIdx.x;
  if (i < n) out[i] = f2bf(in[i]);
}

// ---------------- transpose + convert: in[R][C] f32 -> out[C][R] bf16 ----------------
__global__ __launch_bounds__(256) void k_transpose_bf(const float* __restrict__ in,
                                                      u16* __restrict__ out,
                                                      int R, int C) {
  __shared__ float tile[32][33];
  int tx = threadIdx.x, ty = threadIdx.y;          // 32 x 8
  int col = blockIdx.x * 32 + tx;
#pragma unroll
  for (int j = ty; j < 32; j += 8) {
    int row = blockIdx.y * 32 + j;
    tile[j][tx] = in[(size_t)row * C + col];
  }
  __syncthreads();
  int orow = blockIdx.y * 32 + tx;                 // original row
#pragma unroll
  for (int j = ty; j < 32; j += 8) {
    int ocol = blockIdx.x * 32 + j;                // original col
    out[(size_t)ocol * R + orow] = f2bf(tile[tx][j]);
  }
}

// ---------------- shared GEMM core: 16x64 per wave, B tile async-staged in LDS ----
// xrow: this wave's A rows (bf16, stride EDIM). WTsrc: B^T [cols][1024] bf16.
// Bsm: __shared__ u16[2][64*LDST]. 16 chunks of K=64, double buffered.
__device__ __forceinline__ void gemm_acc_async(const u16* __restrict__ xrow,
                                               const u16* __restrict__ WTsrc,
                                               int col0, int tid, int n, int hi,
                                               u16* Bsm, v8f acc[4]) {
  const int NCH = EDIM / 64;                 // 16 K-chunks
  // Stage chunk kc into buffer buf: 64 rows x 64 u16 = 512 x 16B pieces,
  // 128 threads x 4 pieces.
  auto issue = [&](int kc, int buf) {
#pragma unroll
    for (int i = 0; i < 4; ++i) {
      int p = tid + i * 128;
      int r = p >> 3;                        // B row (output col)
      int o = (p & 7) * 8;                   // u16 offset within row
      u32 loff = (u32)(size_t)&Bsm[buf * 64 * LDST + r * LDST + o];
      async_ld_b128(loff, WTsrc + (size_t)(col0 + r) * EDIM + kc + o);
    }
  };
  issue(0, 0);
  for (int c = 0; c < NCH; ++c) {
    wait_async0();                           // own async DMAs landed in LDS
    __syncthreads();                         // all waves' DMAs visible
    if (c + 1 < NCH) issue((c + 1) * 64, (c + 1) & 1);  // prefetch next chunk

    const u16* bb = Bsm + (c & 1) * 64 * LDST;
    int kb = c * 64;
    VB a0, a1;                               // A-frags for both 32-K steps
    a0.q[0] = ld16(xrow + kb + hi * 16);      a0.q[1] = ld16(xrow + kb + hi * 16 + 8);
    a1.q[0] = ld16(xrow + kb + 32 + hi * 16); a1.q[1] = ld16(xrow + kb + 40 + hi * 16);
#pragma unroll
    for (int t = 0; t < 4; ++t) {
      const u16* w0 = bb + (t * 16 + n) * LDST + hi * 16;
      VB b0; b0.q[0] = ld16(w0);      b0.q[1] = ld16(w0 + 8);
      acc[t] = wmma_bf16(a0, b0, acc[t]);
      VB b1; b1.q[0] = ld16(w0 + 32); b1.q[1] = ld16(w0 + 40);
      acc[t] = wmma_bf16(a1, b1, acc[t]);
    }
  }
}

// ---------------- QKV GEMM: Xb(8192x1024) @ Wqkv -> Q/K row-major, V transposed ----
__global__ __launch_bounds__(128) void k_qkv(const u16* __restrict__ Xb,
                                             const u16* __restrict__ WT,
                                             u16* __restrict__ Qh,
                                             u16* __restrict__ Kh,
                                             u16* __restrict__ Vt) {
  __shared__ u16 Bsm[2][64 * LDST];
  int tid = threadIdx.x;
  int wid = tid >> 5, lane = tid & 31;
  int n = lane & 15, hi = lane >> 4;
  int blkRow = blockIdx.x / 48;              // 0..127
  int blkCol = blockIdx.x % 48;              // 0..47
  int row0 = blkRow * 64 + wid * 16;
  int col0 = blkCol * 64;

  const u16* xrow = Xb + (size_t)(row0 + n) * EDIM;
  v8f acc[4];
#pragma unroll
  for (int t = 0; t < 4; ++t) acc[t] = vzero();

  gemm_acc_async(xrow, WT, col0, tid, n, hi, &Bsm[0][0], acc);

  int b  = row0 >> 11;
  int l0 = row0 & 2047;
#pragma unroll
  for (int t = 0; t < 4; ++t) {
    int j  = col0 + t * 16 + n;
    int h  = j / (3 * ADIM);
    int tt = j % (3 * ADIM);
    size_t bh = (size_t)b * HDIM + h;
#pragma unroll
    for (int r = 0; r < 8; ++r) {
      int l = l0 + r + 8 * hi;
      float v = acc[t][r];
      if (tt < ADIM) {
        Qh[(bh * LDIM + l) * ADIM + tt] = f2bf(v * 0.125f);     // fold 1/sqrt(64)
      } else if (tt < 2 * ADIM) {
        Kh[(bh * LDIM + l) * ADIM + (tt - ADIM)] = f2bf(v);
      } else {
        Vt[(bh * ADIM + (tt - 2 * ADIM)) * LDIM + l] = f2bf(v); // transposed
      }
    }
  }
}

// ---------------- Flash attention, one wave per (b,h,16-row q tile) ----------------
// Computes S^T = K_tile @ Q^T so each lane owns one q-row's scores (split across
// the xor-16 lane pair). P^T repacked into a bf16 B-frag; acc^T = V^T @ P^T.
__global__ __launch_bounds__(128) void k_attn(const u16* __restrict__ Qh,
                                              const u16* __restrict__ Kh,
                                              const u16* __restrict__ Vt,
                                              const unsigned char* __restrict__ mask,
                                              u16* __restrict__ vals) {
  int wave = blockIdx.x * 4 + (threadIdx.x >> 5);
  int lane = threadIdx.x & 31;
  int n = lane & 15, hi = lane >> 4;
  int qt = wave & 127;          // L/16 tiles; 4 waves/block share (b,h)
  int bh = wave >> 7;           // 0..63
  int b  = bh >> 4;
  int h  = bh & 15;
  int q0 = qt * 16;

  const u16* qrow = Qh + ((size_t)bh * LDIM + q0 + n) * ADIM;
  VB qb0, qb1;                  // Q^T B-frags (dims 0-31, 32-63)
  qb0.q[0] = ld16(qrow + hi * 8);
  qb0.q[1] = ld16(qrow + 16 + hi * 8);
  qb1.q[0] = ld16(qrow + 32 + hi * 8);
  qb1.q[1] = ld16(qrow + 48 + hi * 8);

  v8f acc[4];
#pragma unroll
  for (int c = 0; c < 4; ++c) acc[c] = vzero();
  float mrow = -1e30f, lrow = 0.f;
  const unsigned char* mptr = mask + ((size_t)b * LDIM + q0 + n) * LDIM;

  for (int k0 = 0; k0 < LDIM; k0 += 32) {
    const u16* krow0 = Kh + ((size_t)bh * LDIM + k0 + n) * ADIM;
    const u16* krow1 = krow0 + 16 * ADIM;
    VB a00, a01, a10, a11;      // K rows as A-frags (two 16-key tiles x two 32-dim halves)
    a00.q[0] = ld16(krow0 + hi * 8);       a00.q[1] = ld16(krow0 + 16 + hi * 8);
    a01.q[0] = ld16(krow0 + 32 + hi * 8);  a01.q[1] = ld16(krow0 + 48 + hi * 8);
    a10.q[0] = ld16(krow1 + hi * 8);       a10.q[1] = ld16(krow1 + 16 + hi * 8);
    a11.q[0] = ld16(krow1 + 32 + hi * 8);  a11.q[1] = ld16(krow1 + 48 + hi * 8);

    v8f t0 = vzero(), t1 = vzero();
    t0 = wmma_bf16(a00, qb0, t0);
    t0 = wmma_bf16(a01, qb1, t0);
    t1 = wmma_bf16(a10, qb0, t1);
    t1 = wmma_bf16(a11, qb1, t1);

    // mask: lane holds keys {hi*8..+7} (t0) and {16+hi*8..+7} (t1) of row q0+n
    u64 mk0 = *reinterpret_cast<const u64*>(mptr + k0 + hi * 8);
    u64 mk1 = *reinterpret_cast<const u64*>(mptr + k0 + 16 + hi * 8);

    float s0[8], s1[8];
#pragma unroll
    for (int r = 0; r < 8; ++r) {
      s0[r] = ((mk0 >> (8 * r)) & 0xffull) ? -1e30f : t0[r];
      s1[r] = ((mk1 >> (8 * r)) & 0xffull) ? -1e30f : t1[r];
    }
    float mx = s0[0];
#pragma unroll
    for (int r = 0; r < 8; ++r) { mx = fmaxf(mx, s0[r]); mx = fmaxf(mx, s1[r]); }
    mx = fmaxf(mx, __shfl_xor(mx, 16));
    float mnew  = fmaxf(mrow, mx);
    float alpha = __expf(mrow - mnew);
    mrow = mnew;

    float p0[8], p1[8], ssum = 0.f;
#pragma unroll
    for (int r = 0; r < 8; ++r) {
      p0[r] = __expf(s0[r] - mnew);
      p1[r] = __expf(s1[r] - mnew);
      ssum += p0[r] + p1[r];
    }
    ssum += __shfl_xor(ssum, 16);
    lrow = lrow * alpha + ssum;
#pragma unroll
    for (int c = 0; c < 4; ++c)
#pragma unroll
      for (int r = 0; r < 8; ++r) acc[c][r] *= alpha;

    // Repack P^T into B-frag: lane<16 needs keys 0..15 = {own t0, partner t0};
    // lane>=16 needs keys 16..31 = {partner t1, own t1}. One xor-16 exchange.
    float xa[8], xb[8];
#pragma unroll
    for (int r = 0; r < 8; ++r) {
      float sx = __shfl_xor(p0[r], 16);
      float sy = __shfl_xor(p1[r], 16);
      xa[r] = hi ? sy : p0[r];
      xb[r] = hi ? p1[r] : sx;
    }
    VB pb;
#pragma unroll
    for (int i = 0; i < 4; ++i) {
      pb.u[i]     = pack2(xa[2 * i], xa[2 * i + 1]);
      pb.u[4 + i] = pack2(xb[2 * i], xb[2 * i + 1]);
    }

#pragma unroll
    for (int c = 0; c < 4; ++c) {           // V^T A-frags, 4 chunks of 16 dims
      const u16* vrow = Vt + ((size_t)bh * ADIM + c * 16 + n) * LDIM + k0;
      VB va;
      va.q[0] = ld16(vrow + hi * 16);
      va.q[1] = ld16(vrow + hi * 16 + 8);
      acc[c] = wmma_bf16(va, pb, acc[c]);
    }
  }

  // fully-masked rows: running max never left -1e30 -> output zeros (matches ref)
  float rcp = (mrow > -1e29f && lrow > 0.f) ? (1.0f / lrow) : 0.f;
  u16* orow = vals + ((size_t)b * LDIM + q0 + n) * EDIM + h * ADIM;
#pragma unroll
  for (int c = 0; c < 4; ++c) {
    uint4 st;
    st.x = pack2(acc[c][0] * rcp, acc[c][1] * rcp);
    st.y = pack2(acc[c][2] * rcp, acc[c][3] * rcp);
    st.z = pack2(acc[c][4] * rcp, acc[c][5] * rcp);
    st.w = pack2(acc[c][6] * rcp, acc[c][7] * rcp);
    *reinterpret_cast<uint4*>(orow + c * 16 + hi * 8) = st;
  }
}

// ---------------- Output projection: vals(8192x1024) @ Wout -> fp32 out ----------------
__global__ __launch_bounds__(128) void k_out(const u16* __restrict__ vals,
                                             const u16* __restrict__ WoT,
                                             float* __restrict__ out) {
  __shared__ u16 Bsm[2][64 * LDST];
  int tid = threadIdx.x;
  int wid = tid >> 5, lane = tid & 31;
  int n = lane & 15, hi = lane >> 4;
  int blkRow = blockIdx.x / 16;              // 0..127
  int blkCol = blockIdx.x % 16;              // 0..15
  int row0 = blkRow * 64 + wid * 16;
  int col0 = blkCol * 64;

  const u16* xrow = vals + (size_t)(row0 + n) * EDIM;
  v8f acc[4];
#pragma unroll
  for (int t = 0; t < 4; ++t) acc[t] = vzero();

  gemm_acc_async(xrow, WoT, col0, tid, n, hi, &Bsm[0][0], acc);

#pragma unroll
  for (int t = 0; t < 4; ++t)
#pragma unroll
    for (int r = 0; r < 8; ++r)
      out[(size_t)(row0 + r + 8 * hi) * EDIM + col0 + t * 16 + n] = acc[t][r];
}

extern "C" void kernel_launch(void* const* d_in, const int* in_sizes, int n_in,
                              void* d_out, int out_size, void* d_ws, size_t ws_size,
                              hipStream_t stream) {
  const float* emb          = (const float*)d_in[0];
  const unsigned char* mask = (const unsigned char*)d_in[1];   // bool array
  const float* Wqkv         = (const float*)d_in[2];
  const float* Wout         = (const float*)d_in[3];
  float* out                = (float*)d_out;

  char* ws = (char*)d_ws;
  const size_t MB = 1024 * 1024;
  u16* Xb   = (u16*)(ws + 0 * MB);    // 16 MiB  bf16 embeddings
  u16* WqT  = (u16*)(ws + 16 * MB);   //  6 MiB  W_qkv^T bf16 [3072][1024]
  u16* WoT  = (u16*)(ws + 22 * MB);   //  2 MiB  W_out^T bf16 [1024][1024]
  u16* Qh   = (u16*)(ws + 24 * MB);   // 16 MiB  [B*H][L][64] (pre-scaled)
  u16* Kh   = (u16*)(ws + 40 * MB);   // 16 MiB  [B*H][L][64]
  u16* Vt   = (u16*)(ws + 56 * MB);   // 16 MiB  [B*H][64][L]
  u16* vals = (u16*)(ws + 72 * MB);   // 16 MiB  attention output, [B*L][E]
  // total 88 MiB of d_ws used

  int nX = BDIM * LDIM * EDIM;
  k_f2bf<<<nX / 256, 256, 0, stream>>>(emb, Xb, nX);
  k_transpose_bf<<<dim3(E3 / 32, EDIM / 32), dim3(32, 8), 0, stream>>>(Wqkv, WqT, EDIM, E3);
  k_transpose_bf<<<dim3(EDIM / 32, EDIM / 32), dim3(32, 8), 0, stream>>>(Wout, WoT, EDIM, EDIM);
  k_qkv<<<128 * 48, 128, 0, stream>>>(Xb, WqT, Qh, Kh, Vt);
  k_attn<<<(BDIM * HDIM * (LDIM / 16)) / 4, 128, 0, stream>>>(Qh, Kh, Vt, mask, vals);
  k_out<<<128 * 16, 128, 0, stream>>>(vals, WoT, out);
}
